// WinCLIP_43825846288754
// MI455X (gfx1250) — matrix operand order repaired and checked by reading, not compiled, not run on hardware
//
#include <hip/hip_runtime.h>
#include <hip/hip_bf16.h>

typedef __attribute__((ext_vector_type(16))) __bf16 v16bf;
typedef __attribute__((ext_vector_type(8)))  __bf16 v8bf;
typedef __attribute__((ext_vector_type(8)))  float  v8f;

#define TEMP      0.07f
#define DIM       640
#define BATCH     512
#define GRID      15
#define NPATCH    225            // 15*15
#define NW2       196            // (15-1)^2
#define NW3       169            // (15-2)^2
#define NREF      450            // K=2 * 225
#define NREF_PAD  480            // 30 tiles of 16 (15 tile-pairs)
#define MROWS     (BATCH*NPATCH) // 115200
#define MBLK      128            // M rows per workgroup
#define KSTEPS    (DIM/32)       // 20
#define NTILES    (NREF_PAD/16)  // 30

__device__ __forceinline__ unsigned short f2bf(float f) {
  unsigned u = __float_as_uint(f);
  u += 0x7FFFu + ((u >> 16) & 1u);   // round-to-nearest-even
  return (unsigned short)(u >> 16);
}

__device__ __forceinline__ float wave_red_add(float v) {
#pragma unroll
  for (int m = 16; m >= 1; m >>= 1) v += __shfl_xor(v, m, 32);
  return v;
}

// ---------------------------------------------------------------------------
// u = t1/||t1|| - t0/||t0||   (single wave)
// ---------------------------------------------------------------------------
__global__ void prep_text_kernel(const float* __restrict__ text,
                                 float* __restrict__ u) {
  const int lane = threadIdx.x & 31;
  float4 t0[5], t1[5];
  float ss0 = 0.f, ss1 = 0.f;
#pragma unroll
  for (int i = 0; i < 5; ++i) {
    t0[i] = *(const float4*)(text + i * 128 + lane * 4);
    t1[i] = *(const float4*)(text + DIM + i * 128 + lane * 4);
    ss0 += t0[i].x * t0[i].x + t0[i].y * t0[i].y + t0[i].z * t0[i].z + t0[i].w * t0[i].w;
    ss1 += t1[i].x * t1[i].x + t1[i].y * t1[i].y + t1[i].z * t1[i].z + t1[i].w * t1[i].w;
  }
  ss0 = wave_red_add(ss0);
  ss1 = wave_red_add(ss1);
  const float rn0 = 1.f / fmaxf(sqrtf(ss0), 1e-12f);
  const float rn1 = 1.f / fmaxf(sqrtf(ss1), 1e-12f);
#pragma unroll
  for (int i = 0; i < 5; ++i) {
    float4 o;
    o.x = t1[i].x * rn1 - t0[i].x * rn0;
    o.y = t1[i].y * rn1 - t0[i].y * rn0;
    o.z = t1[i].z * rn1 - t0[i].z * rn0;
    o.w = t1[i].w * rn1 - t0[i].w * rn0;
    *(float4*)(u + i * 128 + lane * 4) = o;
  }
}

// ---------------------------------------------------------------------------
// One wave per row: score = sigmoid(dot(e,u)/(||e||*T)); optional 1/||e|| out.
// ---------------------------------------------------------------------------
__global__ void row_score_kernel(const float* __restrict__ emb,
                                 const float* __restrict__ u,
                                 float* __restrict__ scores,
                                 float* __restrict__ rnormOut,
                                 int rows) {
  const int gw   = (int)((blockIdx.x * blockDim.x + threadIdx.x) >> 5);
  const int lane = threadIdx.x & 31;
  if (gw >= rows) return;
  const float* e = emb + (size_t)gw * DIM;
  float ss = 0.f, du = 0.f;
#pragma unroll
  for (int i = 0; i < 5; ++i) {
    float4 v = *(const float4*)(e + i * 128 + lane * 4);
    float4 w = *(const float4*)(u + i * 128 + lane * 4);
    ss += v.x * v.x + v.y * v.y + v.z * v.z + v.w * v.w;
    du += v.x * w.x + v.y * w.y + v.z * w.z + v.w * w.w;
  }
  ss = wave_red_add(ss);
  du = wave_red_add(du);
  const float norm  = fmaxf(sqrtf(ss), 1e-12f);
  const float score = 1.f / (1.f + __expf(-du / (norm * TEMP)));
  if (lane == 0) {
    scores[gw] = score;
    if (rnormOut) rnormOut[gw] = 1.f / norm;
  }
}

// ---------------------------------------------------------------------------
// Normalize ref rows -> bf16 (row-major, padded to NREF_PAD rows).
// ---------------------------------------------------------------------------
__global__ void prep_ref_kernel(const float* __restrict__ ref,
                                unsigned short* __restrict__ refs16) {
  const int gw   = (int)((blockIdx.x * blockDim.x + threadIdx.x) >> 5);
  const int lane = threadIdx.x & 31;
  if (gw >= NREF_PAD) return;
  unsigned short* out = refs16 + (size_t)gw * DIM;
  if (gw >= NREF) {
#pragma unroll
    for (int i = 0; i < 5; ++i)
      *(ushort4*)(out + i * 128 + lane * 4) = make_ushort4(0, 0, 0, 0);
    return;
  }
  const float* e = ref + (size_t)gw * DIM;
  float4 v[5];
  float ss = 0.f;
#pragma unroll
  for (int i = 0; i < 5; ++i) {
    v[i] = *(const float4*)(e + i * 128 + lane * 4);
    ss += v[i].x * v[i].x + v[i].y * v[i].y + v[i].z * v[i].z + v[i].w * v[i].w;
  }
  ss = wave_red_add(ss);
  const float rn = 1.f / fmaxf(sqrtf(ss), 1e-12f);
#pragma unroll
  for (int i = 0; i < 5; ++i)
    *(ushort4*)(out + i * 128 + lane * 4) =
        make_ushort4(f2bf(v[i].x * rn), f2bf(v[i].y * rn),
                     f2bf(v[i].z * rn), f2bf(v[i].w * rn));
}

// ---------------------------------------------------------------------------
// assoc[m] = (1 - max_n sim[m,n]) / 2   via bf16 WMMA (M=115200, N=450, K=640)
// Each block: 128-row A panel staged (normalized fp32->bf16) in 160 KB LDS.
// Each of 8 waves owns 16 rows; processes N-tile PAIRS (two accumulators) so
// every A fragment feeds two v_wmma_f32_16x16x32_bf16 ops.
// ---------------------------------------------------------------------------
__global__ __launch_bounds__(256) void assoc_gemm_kernel(
    const float* __restrict__ patches,        // (MROWS, DIM) fp32
    const float* __restrict__ rnorm,          // (MROWS)
    const unsigned short* __restrict__ refs,  // (NREF_PAD, DIM) bf16
    float* __restrict__ assoc) {              // (MROWS)
  extern __shared__ unsigned short As[];      // MBLK * DIM bf16 = 160 KB
  const int t     = threadIdx.x;
  const int mbase = blockIdx.x * MBLK;

  // Stage + normalize A panel into LDS (bf16).
  for (int f = t; f < MBLK * (DIM / 4); f += 256) {
    const int row = f / (DIM / 4);
    const int kc  = (f - row * (DIM / 4)) * 4;
    float4 v = *(const float4*)(patches + (size_t)(mbase + row) * DIM + kc);
    const float rn = rnorm[mbase + row];
    *(ushort4*)(As + row * DIM + kc) =
        make_ushort4(f2bf(v.x * rn), f2bf(v.y * rn),
                     f2bf(v.z * rn), f2bf(v.w * rn));
  }
  __syncthreads();

  const int lane = t & 31;
  const int w    = t >> 5;        // wave id: owns rows w*16 .. w*16+15
  const int nl   = lane & 15;
  const int half = lane >> 4;

  const __bf16* Ab   = (const __bf16*)As;
  // A fragment: lane holds row (lane&15); element e maps
  // to k = 16*half + e  ->  two contiguous 8-elem chunks.
  const __bf16* arow = Ab + (w * 16 + nl) * DIM + half * 16;
  // B fragment: lane holds column n = nt*16 + (lane&15), k = kt*32 + half*16 + e.
  const __bf16* bcol = (const __bf16*)refs + half * 16;

  float runmax[8];
#pragma unroll
  for (int v = 0; v < 8; ++v) runmax[v] = -3.0e38f;

  for (int nt = 0; nt < NTILES; nt += 2) {
    v8f c0 = {0.f, 0.f, 0.f, 0.f, 0.f, 0.f, 0.f, 0.f};
    v8f c1 = {0.f, 0.f, 0.f, 0.f, 0.f, 0.f, 0.f, 0.f};
    const __bf16* brow0 = bcol + (size_t)(nt * 16 + nl) * DIM;
    const __bf16* brow1 = brow0 + (size_t)16 * DIM;
#pragma unroll 4
    for (int kt = 0; kt < KSTEPS; ++kt) {
      v8bf a0 = *(const v8bf*)(arow + kt * 32);
      v8bf a1 = *(const v8bf*)(arow + kt * 32 + 8);
      v16bf a = __builtin_shufflevector(a0, a1, 0, 1, 2, 3, 4, 5, 6, 7,
                                        8, 9, 10, 11, 12, 13, 14, 15);
      v16bf b0 = *(const v16bf*)(brow0 + kt * 32);
      v16bf b1 = *(const v16bf*)(brow1 + kt * 32);
      c0 = __builtin_amdgcn_wmma_f32_16x16x32_bf16(
          false, a, false, b0, (short)0, c0, false, false);
      c1 = __builtin_amdgcn_wmma_f32_16x16x32_bf16(
          false, a, false, b1, (short)0, c1, false, false);
    }
    if (nt * 16 + nl < NREF) {            // mask padded N columns (tile nt)
#pragma unroll
      for (int v = 0; v < 8; ++v) runmax[v] = fmaxf(runmax[v], c0[v]);
    }
    if ((nt + 1) * 16 + nl < NREF) {      // mask padded N columns (tile nt+1)
#pragma unroll
      for (int v = 0; v < 8; ++v) runmax[v] = fmaxf(runmax[v], c1[v]);
    }
  }

  // Reduce max over the 16 columns inside each 16-lane half
  // (C layout: M = v + 8*half, N = lane&15).
#pragma unroll
  for (int s = 1; s < 16; s <<= 1) {
#pragma unroll
    for (int v = 0; v < 8; ++v)
      runmax[v] = fmaxf(runmax[v], __shfl_xor(runmax[v], s, 32));
  }
  if (nl == 0) {
#pragma unroll
    for (int v = 0; v < 8; ++v)
      assoc[mbase + w * 16 + half * 8 + v] = 0.5f * (1.0f - runmax[v]);
  }
}

// ---------------------------------------------------------------------------
// Harmonic aggregation of both window scales (analytic coverage), 3-way
// harmonic mean with patch scores, combine with visual association.
// ---------------------------------------------------------------------------
__global__ void combine_kernel(const float* __restrict__ s0,
                               const float* __restrict__ s1,
                               const float* __restrict__ sp,
                               const float* __restrict__ assoc,
                               float* __restrict__ out) {
  const int idx = blockIdx.x * blockDim.x + threadIdx.x;
  if (idx >= MROWS) return;
  const int b = idx / NPATCH;
  const int p = idx - b * NPATCH;
  const int y = p / GRID, x = p - (p / GRID) * GRID;

  // scale 2: 14x14 windows; window (i,j) covers patch iff i<=y<=i+1, j<=x<=j+1
  float inv = 0.f; int cnt = 0;
  {
    const int i0 = max(y - 1, 0), i1 = min(y, 13);
    const int j0 = max(x - 1, 0), j1 = min(x, 13);
    for (int i = i0; i <= i1; ++i)
      for (int j = j0; j <= j1; ++j) {
        inv += 1.f / fmaxf(s0[b * NW2 + i * 14 + j], 1e-8f);
        ++cnt;
      }
  }
  const float m0 = (float)cnt / inv;

  // scale 3: 13x13 windows
  inv = 0.f; cnt = 0;
  {
    const int i0 = max(y - 2, 0), i1 = min(y, 12);
    const int j0 = max(x - 2, 0), j1 = min(x, 12);
    for (int i = i0; i <= i1; ++i)
      for (int j = j0; j <= j1; ++j) {
        inv += 1.f / fmaxf(s1[b * NW3 + i * 13 + j], 1e-8f);
        ++cnt;
      }
  }
  const float m1 = (float)cnt / inv;

  const float m2 = sp[idx];
  const float hm = 3.f / (1.f / fmaxf(m0, 1e-8f) +
                          1.f / fmaxf(m1, 1e-8f) +
                          1.f / fmaxf(m2, 1e-8f));
  out[idx] = 0.5f * (hm + assoc[idx]);
}

// ---------------------------------------------------------------------------
extern "C" void kernel_launch(void* const* d_in, const int* in_sizes, int n_in,
                              void* d_out, int out_size, void* d_ws, size_t ws_size,
                              hipStream_t stream) {
  const float* win0  = (const float*)d_in[0];  // (512,196,640)
  const float* win1  = (const float*)d_in[1];  // (512,169,640)
  const float* patch = (const float*)d_in[2];  // (512,225,640)
  const float* refs  = (const float*)d_in[3];  // (2,225,640)
  const float* text  = (const float*)d_in[4];  // (2,640)
  (void)in_sizes; (void)n_in; (void)out_size; (void)ws_size;

  char* ws = (char*)d_ws;
  size_t o = 0;
  float* u = (float*)(ws + o);                 o = 4096;
  unsigned short* refs16 = (unsigned short*)(ws + o);
  o += (size_t)NREF_PAD * DIM * 2;             o = (o + 255) & ~(size_t)255;
  float* rnormP = (float*)(ws + o);            o += (size_t)MROWS * 4;  o = (o + 255) & ~(size_t)255;
  float* sc0    = (float*)(ws + o);            o += (size_t)BATCH * NW2 * 4;  o = (o + 255) & ~(size_t)255;
  float* sc1    = (float*)(ws + o);            o += (size_t)BATCH * NW3 * 4;  o = (o + 255) & ~(size_t)255;
  float* scp    = (float*)(ws + o);            o += (size_t)MROWS * 4;  o = (o + 255) & ~(size_t)255;
  float* assocB = (float*)(ws + o);            o += (size_t)MROWS * 4;

  prep_text_kernel<<<1, 32, 0, stream>>>(text, u);

  const int rows0 = BATCH * NW2;
  row_score_kernel<<<(rows0 * 32 + 255) / 256, 256, 0, stream>>>(win0, u, sc0, nullptr, rows0);
  const int rows1 = BATCH * NW3;
  row_score_kernel<<<(rows1 * 32 + 255) / 256, 256, 0, stream>>>(win1, u, sc1, nullptr, rows1);
  row_score_kernel<<<(MROWS * 32 + 255) / 256, 256, 0, stream>>>(patch, u, scp, rnormP, MROWS);

  prep_ref_kernel<<<(NREF_PAD * 32 + 255) / 256, 256, 0, stream>>>(refs, refs16);

  assoc_gemm_kernel<<<MROWS / MBLK, 256, MBLK * DIM * 2, stream>>>(
      patch, rnormP, refs16, assocB);

  combine_kernel<<<(MROWS + 255) / 256, 256, 0, stream>>>(sc0, sc1, scp, assocB,
                                                          (float*)d_out);
}
